// SFMamba_3D_12000138625321
// MI455X (gfx1250) — matrix operand order repaired and checked by reading, compiled, not run on hardware
//
#include <hip/hip_runtime.h>
#include <math.h>

// ---------------- problem constants (fixed by the reference) ----------------
#define B0     2
#define NFR    8
#define DIMC   384
#define HH     32
#define WW     32
#define HWSZ   1024              // HH*WW
#define LSEQ   8192              // NFR*HWSZ
#define LTOT   16384             // B0*LSEQ
#define DINNER 384
#define DTRANK 24
#define DPROJ  26                // DTRANK + 2*D_STATE
#define HID    1536
#define SQH    0.7071067811865476f
#define S3C    0.3535533905932738f   // (1/sqrt(2))^3

typedef __attribute__((ext_vector_type(16))) __bf16 v16bf;
typedef __attribute__((ext_vector_type(8)))  float  v8f;

__device__ __forceinline__ unsigned short f2bf(float f) {
  unsigned int u = __float_as_uint(f);
  u += 0x7FFFu + ((u >> 16) & 1u);            // round-to-nearest-even
  return (unsigned short)(u >> 16);
}
__device__ __forceinline__ float bf2f(unsigned short h) {
  return __uint_as_float(((unsigned int)h) << 16);
}
__device__ __forceinline__ float siluf(float v) { return v / (1.f + __expf(-v)); }
__device__ __forceinline__ float softplusf(float v) { return v > 20.f ? v : log1pf(__expf(v)); }

// ---------------- generic elementwise ----------------
__global__ void k_f32_to_bf16(const float* __restrict__ in, unsigned short* __restrict__ out, int n) {
  int i = blockIdx.x * blockDim.x + threadIdx.x;
  if (i < n) out[i] = f2bf(in[i]);
}

// x (B,NF,C,H,W) -> xf (B,L,C)
__global__ void k_pack_xf(const float* __restrict__ x, float* __restrict__ xf) {
  int i = blockIdx.x * blockDim.x + threadIdx.x;
  if (i >= LTOT * DIMC) return;
  int c = i % DIMC; int l = (i / DIMC) % LSEQ; int b = i / (DIMC * LSEQ);
  int f = l >> 10; int hw = l & 1023;
  xf[i] = x[(((size_t)(b * NFR + f) * DIMC + c) << 10) + hw];
}

// LayerNorm over C=384, bf16 output. One block (128 thr) per row.
__global__ __launch_bounds__(128) void k_layernorm_bf16(
    const float* __restrict__ xin, const float* __restrict__ g,
    const float* __restrict__ bta, unsigned short* __restrict__ outh) {
  int row = blockIdx.x, t = threadIdx.x;
  const float* xr = xin + (size_t)row * DIMC;
  __shared__ float red[128];
  float s = 0.f;
  for (int c = t; c < DIMC; c += 128) s += xr[c];
  red[t] = s; __syncthreads();
  for (int o = 64; o > 0; o >>= 1) { if (t < o) red[t] += red[t + o]; __syncthreads(); }
  float mu = red[0] * (1.f / DIMC);
  __syncthreads();
  float v = 0.f;
  for (int c = t; c < DIMC; c += 128) { float d = xr[c] - mu; v += d * d; }
  red[t] = v; __syncthreads();
  for (int o = 64; o > 0; o >>= 1) { if (t < o) red[t] += red[t + o]; __syncthreads(); }
  float rs = rsqrtf(red[0] * (1.f / DIMC) + 1e-5f);
  unsigned short* orow = outh + (size_t)row * DIMC;
  for (int c = t; c < DIMC; c += 128) orow[c] = f2bf((xr[c] - mu) * rs * g[c] + bta[c]);
}

// ---------------- WMMA bf16 GEMM: D = A(MxK) * W(NxK)^T ----------------
// mode bit0: bf16 output; bit2: final scatter to (B,NF,C,H,W) + residual
__global__ __launch_bounds__(256) void k_gemm_bf16(
    const unsigned short* __restrict__ A, const unsigned short* __restrict__ Wm,
    const float* __restrict__ bias, const float* __restrict__ addp,
    float* __restrict__ outf, unsigned short* __restrict__ outh,
    const float* __restrict__ resid, int M, int N, int K, int mode) {
  int wid  = blockIdx.x * (blockDim.x >> 5) + (threadIdx.x >> 5);
  int lane = threadIdx.x & 31;
  int tilesN = N >> 4;
  int total  = (M >> 4) * tilesN;
  if (wid >= total) return;
  int tmi = wid / tilesN;
  int tm  = tmi << 4;
  int tn  = (wid - tmi * tilesN) << 4;
  int half = lane >> 4;
  int l15  = lane & 15;
  // A fragment: lanes 0-15 carry K {0..7,16..23}, lanes 16-31 K {8..15,24..31}
  const unsigned short* ap = A  + (size_t)(tm + l15) * K + (half << 3);
  // B fragment: N = lane&15; lanes 0-15 K {0..7,16..23}, lanes 16-31 K {8..15,24..31}
  const unsigned short* bp = Wm + (size_t)(tn + l15) * K + (half << 3);
  v8f acc = {};
  union FragT { uint4 u[2]; v16bf v; };
  for (int k0 = 0; k0 < K; k0 += 32) {
    FragT fa, fb;
    fa.u[0] = *(const uint4*)(ap);
    fa.u[1] = *(const uint4*)(ap + 16);
    fb.u[0] = *(const uint4*)(bp);
    fb.u[1] = *(const uint4*)(bp + 16);
    __builtin_prefetch(ap + 32, 0, 1);
    __builtin_prefetch(bp + 32, 0, 1);
    acc = __builtin_amdgcn_wmma_f32_16x16x32_bf16(false, fa.v, false, fb.v,
                                                  (short)0, acc, false, false);
    ap += 32; bp += 32;
  }
  int n = tn + l15;
  float bv = bias ? bias[n] : 0.f;
#pragma unroll
  for (int r = 0; r < 8; ++r) {
    int m = tm + r + (half << 3);
    float v = acc[r] + bv;
    if (mode & 4) {                       // final: scatter + residual
      int b = m >> 13;                    // / LSEQ
      int l = m & (LSEQ - 1);
      int f = l >> 10; int hw = l & 1023;
      size_t off = (((size_t)(b * NFR + f) * DIMC + n) << 10) + hw;
      outf[off] = v + resid[off];
    } else {
      size_t off = (size_t)m * N + n;
      if (addp) v += addp[off];
      if (mode & 1) outh[off] = f2bf(v);
      else          outf[off] = v;
    }
  }
}

// ---------------- mamba pieces ----------------
// causal depthwise conv1d (k=4, pad-left 3) on x-half of xz (B,L,768) + silu
__global__ void k_conv1d_silu(const float* __restrict__ xz, const float* __restrict__ cw,
                              const float* __restrict__ cb, float* __restrict__ xc, int rev) {
  int i = blockIdx.x * blockDim.x + threadIdx.x;
  if (i >= LTOT * DINNER) return;
  int d = i % DINNER; int l = (i / DINNER) % LSEQ; int b = i / (DINNER * LSEQ);
  float acc = cb[d];
#pragma unroll
  for (int j = 0; j < 4; ++j) {
    int ls = l - 3 + j;
    if (ls >= 0) {
      int src = rev ? (LSEQ - 1 - ls) : ls;
      acc += cw[d * 4 + j] * xz[((size_t)b * LSEQ + src) * (2 * DINNER) + d];
    }
  }
  xc[i] = siluf(acc);
}

// dbl = xc @ x_proj_w^T  (26 outputs, K=384)
__global__ void k_xproj(const float* __restrict__ xc, const float* __restrict__ xpw,
                        float* __restrict__ dbl) {
  int i = blockIdx.x * blockDim.x + threadIdx.x;
  if (i >= LTOT * DPROJ) return;
  int o = i % DPROJ; int row = i / DPROJ;
  const float* xr = xc + (size_t)row * DINNER;
  const float* wr = xpw + o * DINNER;
  float acc = 0.f;
  for (int k = 0; k < DINNER; ++k) acc += xr[k] * wr[k];
  dbl[i] = acc;
}

// dt = softplus(dbl[:,:24] @ dt_w^T + dt_b); dA = exp(dt*(-exp(A_log))); dBx = dt*Bm*xt
__global__ void k_dt(const float* __restrict__ dbl, const float* __restrict__ dtw,
                     const float* __restrict__ dtb, const float* __restrict__ alog,
                     const float* __restrict__ xc, float* __restrict__ dA,
                     float* __restrict__ dBx) {
  int i = blockIdx.x * blockDim.x + threadIdx.x;
  if (i >= LTOT * DINNER) return;
  int d = i % DINNER; int row = i / DINNER;
  const float* dr = dbl + (size_t)row * DPROJ;
  float acc = dtb[d];
#pragma unroll
  for (int r = 0; r < DTRANK; ++r) acc += dr[r] * dtw[d * DTRANK + r];
  float dt = softplusf(acc);
  float a  = -__expf(alog[d]);
  dA[i]  = __expf(dt * a);
  dBx[i] = dt * dr[DTRANK] * xc[i];
}

// block-parallel first-order linear recurrence h=dA*h+dBx over L, fused output
__global__ __launch_bounds__(256) void k_scan(
    const float* __restrict__ dA, const float* __restrict__ dBx,
    const float* __restrict__ dbl, const float* __restrict__ xc,
    const float* __restrict__ xz, const float* __restrict__ Dp,
    float* __restrict__ ys, int rev) {
  int bd = blockIdx.x; int b = bd / DINNER; int d = bd % DINNER;
  int t = threadIdx.x;
  const int SEG = LSEQ / 256;   // 32
  __shared__ float sa[256], sb[256];
  size_t base = ((size_t)b * LSEQ) * DINNER + d;
  int l0 = t * SEG;
  float a = 1.f, s = 0.f;
  for (int i2 = 0; i2 < SEG; ++i2) {
    size_t idx = base + (size_t)(l0 + i2) * DINNER;
    float da = dA[idx], dbx = dBx[idx];
    s = da * s + dbx; a *= da;
  }
  sa[t] = a; sb[t] = s; __syncthreads();
  for (int off = 1; off < 256; off <<= 1) {
    float ca = sa[t], cb2 = sb[t];
    float pa = 1.f, pb = 0.f;
    if (t >= off) { pa = sa[t - off]; pb = sb[t - off]; }
    __syncthreads();
    if (t >= off) { sa[t] = pa * ca; sb[t] = ca * pb + cb2; }
    __syncthreads();
  }
  float h  = (t == 0) ? 0.f : sb[t - 1];
  float Dv = Dp[d];
  for (int i2 = 0; i2 < SEG; ++i2) {
    int l = l0 + i2;
    size_t idx = base + (size_t)l * DINNER;
    float da = dA[idx], dbx = dBx[idx];
    h = da * h + dbx;
    float cm = dbl[((size_t)b * LSEQ + l) * DPROJ + DTRANK + 1];
    float y  = h * cm + xc[idx] * Dv;
    int zl = rev ? (LSEQ - 1 - l) : l;
    float zv = xz[((size_t)b * LSEQ + zl) * (2 * DINNER) + DINNER + d];
    y *= siluf(zv);
    int lout = rev ? (LSEQ - 1 - l) : l;
    size_t o = ((size_t)b * LSEQ + lout) * DINNER + d;
    if (rev) ys[o] += y; else ys[o] = y;
  }
}

// ---------------- MLP depthwise 3x3x3 conv + exact GELU (bf16 in/out) ----------------
__global__ void k_dwconv_gelu(const unsigned short* __restrict__ hin,
                              const float* __restrict__ dww, const float* __restrict__ dwb,
                              unsigned short* __restrict__ hout) {
  int i = blockIdx.x * blockDim.x + threadIdx.x;
  if (i >= LTOT * HID) return;
  int ch = i % HID; int l = (i / HID) % LSEQ; int b = i / (HID * LSEQ);
  int f = l >> 10, y = (l >> 5) & 31, x = l & 31;
  float acc = dwb[ch];
  const float* wr = dww + ch * 27;
  for (int a2 = 0; a2 < 3; ++a2) {
    int fz = f + a2 - 1; if ((unsigned)fz >= NFR) continue;
    for (int b2 = 0; b2 < 3; ++b2) {
      int yy = y + b2 - 1; if ((unsigned)yy >= HH) continue;
      for (int c2 = 0; c2 < 3; ++c2) {
        int xx = x + c2 - 1; if ((unsigned)xx >= WW) continue;
        int lp = (fz << 10) | (yy << 5) | xx;
        acc += wr[a2 * 9 + b2 * 3 + c2] * bf2f(hin[((size_t)b * LSEQ + lp) * HID + ch]);
      }
    }
  }
  float g = 0.5f * acc * (1.f + erff(acc * SQH));
  hout[i] = f2bf(g);
}

// ---------------- wavelet branch ----------------
// fused dwt3 + gather by scan_index -> xfw (B,L,C)
__global__ void k_dwt_gather(const float* __restrict__ x, const int* __restrict__ sidx,
                             float* __restrict__ xfw) {
  int i = blockIdx.x * blockDim.x + threadIdx.x;
  if (i >= LTOT * DIMC) return;
  int c = i % DIMC; int l = (i / DIMC) % LSEQ; int b = i / (DIMC * LSEQ);
  int p = sidx[l];
  int dz = p >> 10, hy = (p >> 5) & 31, wx = p & 31;
  int dp = dz & 3,  sd = (dz >> 2) & 1;
  int hp = hy & 15, sh = (hy >> 4) & 1;
  int wp = wx & 15, sw = (wx >> 4) & 1;
  float acc = 0.f;
  for (int i3 = 0; i3 < 2; ++i3)
    for (int j = 0; j < 2; ++j)
      for (int k = 0; k < 2; ++k) {
        float v = x[(((size_t)(b * NFR + (2 * dp + i3)) * DIMC + c) << 10) +
                    ((2 * hp + j) << 5) + (2 * wp + k)];
        int sgn = (sd & i3) ^ (sh & j) ^ (sw & k);
        acc += sgn ? -v : v;
      }
  xfw[i] = acc * S3C;
}

__global__ void k_invperm(const int* __restrict__ s, int* __restrict__ inv) {
  int l = blockIdx.x * blockDim.x + threadIdx.x;
  if (l < LSEQ) inv[s[l]] = l;
}

// fused scatter (via inverse perm) + idwt3 -> first half of concat (bf16)
__global__ void k_idwt_cat(const float* __restrict__ om, const int* __restrict__ inv,
                           unsigned short* __restrict__ cat) {
  int i = blockIdx.x * blockDim.x + threadIdx.x;
  if (i >= LTOT * DIMC) return;
  int c = i % DIMC; int q = (i / DIMC) % LSEQ; int b = i / (DIMC * LSEQ);
  int D = q >> 10, Hy = (q >> 5) & 31, Wx = q & 31;
  int dp = D >> 1, i3 = D & 1;
  int hp = Hy >> 1, j = Hy & 1;
  int wp = Wx >> 1, k = Wx & 1;
  float acc = 0.f;
  for (int a2 = 0; a2 < 2; ++a2)
    for (int b2 = 0; b2 < 2; ++b2)
      for (int c2 = 0; c2 < 2; ++c2) {
        int p = (((a2 << 2) + dp) << 10) | (((b2 << 4) + hp) << 5) | ((c2 << 4) + wp);
        int l = inv[p];
        float v = om[((size_t)b * LSEQ + l) * DIMC + c];
        int sgn = (i3 & a2) ^ (j & b2) ^ (k & c2);
        acc += sgn ? -v : v;
      }
  cat[((size_t)b * LSEQ + q) * (2 * DIMC) + c] = f2bf(acc * S3C);
}

__global__ void k_cat_global(const float* __restrict__ og, unsigned short* __restrict__ cat) {
  int i = blockIdx.x * blockDim.x + threadIdx.x;
  if (i >= LTOT * DIMC) return;
  int c = i % DIMC; int row = i / DIMC;
  cat[(size_t)row * (2 * DIMC) + DIMC + c] = f2bf(og[i]);
}

// ---------------- host launch ----------------
extern "C" void kernel_launch(void* const* d_in, const int* in_sizes, int n_in,
                              void* d_out, int out_size, void* d_ws, size_t ws_size,
                              hipStream_t stream) {
  (void)in_sizes; (void)n_in; (void)out_size; (void)ws_size;
#define F(i) ((const float*)d_in[(i)])
  const float* x = F(0);
  const int* sidx = (const int*)d_in[1];
  // input indices: mamba1@2..17, mamba2@18..33, mlp1@34..39, mlp2@40..45,
  // ln g1..b4 @46..53, lin_w@54, lin_b@55

  char* Wp = (char*)d_ws;
  size_t off = 0;
  auto carve = [&](size_t bytes) -> char* {
    char* p = Wp + off; off += (bytes + 255) & ~(size_t)255; return p;
  };
  unsigned short* BW_INW1  = (unsigned short*)carve((size_t)768 * 384 * 2);
  unsigned short* BW_OUTW1 = (unsigned short*)carve((size_t)384 * 384 * 2);
  unsigned short* BW_INW2  = (unsigned short*)carve((size_t)768 * 384 * 2);
  unsigned short* BW_OUTW2 = (unsigned short*)carve((size_t)384 * 384 * 2);
  unsigned short* BW_FC1W1 = (unsigned short*)carve((size_t)1536 * 384 * 2);
  unsigned short* BW_FC2W1 = (unsigned short*)carve((size_t)384 * 1536 * 2);
  unsigned short* BW_FC1W2 = (unsigned short*)carve((size_t)1536 * 384 * 2);
  unsigned short* BW_FC2W2 = (unsigned short*)carve((size_t)384 * 1536 * 2);
  unsigned short* BW_LIN   = (unsigned short*)carve((size_t)384 * 768 * 2);
  float* XF  = (float*)carve((size_t)LTOT * DIMC * 4);
  float* XZ  = (float*)carve((size_t)LTOT * 2 * DINNER * 4);
  float* XC  = (float*)carve((size_t)LTOT * DINNER * 4);
  float* DBL = (float*)carve((size_t)LTOT * DPROJ * 4);
  float* DA  = (float*)carve((size_t)LTOT * DINNER * 4);
  float* DBX = (float*)carve((size_t)LTOT * DINNER * 4);
  float* YS  = (float*)carve((size_t)LTOT * DINNER * 4);
  unsigned short* LNB = (unsigned short*)carve((size_t)LTOT * DIMC * 2);
  float* XM  = (float*)carve((size_t)LTOT * DIMC * 4);
  unsigned short* HB = (unsigned short*)carve((size_t)LTOT * HID * 2);
  unsigned short* BB = (unsigned short*)carve((size_t)LTOT * HID * 2);
  float* OG  = (float*)carve((size_t)LTOT * DIMC * 4);
  float* OM  = (float*)carve((size_t)LTOT * DIMC * 4);
  unsigned short* CAT = (unsigned short*)carve((size_t)LTOT * 2 * DIMC * 2);
  int* IP = (int*)carve((size_t)LSEQ * 4);

  auto g = [](long n) { return dim3((unsigned)((n + 255) / 256)); };
  auto cvt = [&](const float* src, unsigned short* dst, int n) {
    k_f32_to_bf16<<<g(n), 256, 0, stream>>>(src, dst, n);
  };
  auto gemm = [&](const unsigned short* A, const unsigned short* Wm, const float* bias,
                  const float* addp, float* outf, unsigned short* outh, const float* resid,
                  int N, int K, int mode) {
    int tiles = (LTOT / 16) * (N / 16);
    k_gemm_bf16<<<(tiles + 7) / 8, 256, 0, stream>>>(A, Wm, bias, addp, outf, outh, resid,
                                                     LTOT, N, K, mode);
  };

  // weight conversions (every call; no cached state)
  cvt(F(2),  BW_INW1,  768 * 384);  cvt(F(3),  BW_OUTW1, 384 * 384);
  cvt(F(18), BW_INW2,  768 * 384);  cvt(F(19), BW_OUTW2, 384 * 384);
  cvt(F(34), BW_FC1W1, 1536 * 384); cvt(F(38), BW_FC2W1, 384 * 1536);
  cvt(F(40), BW_FC1W2, 1536 * 384); cvt(F(44), BW_FC2W2, 384 * 1536);
  cvt(F(54), BW_LIN,   384 * 768);

  // bidirectional mamba: consumes LN output in LNB, residual xfbuf, writes xm_out
  auto mamba_block = [&](int pbase, const float* xfbuf,
                         const unsigned short* inw, const unsigned short* outw,
                         float* xm_out) {
    gemm(LNB, inw, nullptr, nullptr, XZ, nullptr, nullptr, 2 * DINNER, DIMC, 0);
    for (int dir = 0; dir < 2; ++dir) {
      int o = pbase + 2 + dir * 7;  // conv_w, conv_b, x_proj_w, dt_w, dt_b, A_log, D
      k_conv1d_silu<<<g((long)LTOT * DINNER), 256, 0, stream>>>(XZ, F(o), F(o + 1), XC, dir);
      k_xproj<<<g((long)LTOT * DPROJ), 256, 0, stream>>>(XC, F(o + 2), DBL);
      k_dt<<<g((long)LTOT * DINNER), 256, 0, stream>>>(DBL, F(o + 3), F(o + 4), F(o + 5), XC, DA, DBX);
      k_scan<<<B0 * DINNER, 256, 0, stream>>>(DA, DBX, DBL, XC, XZ, F(o + 6), YS, dir);
    }
    cvt(YS, LNB, LTOT * DINNER);
    gemm(LNB, outw, nullptr, xfbuf, xm_out, nullptr, nullptr, DIMC, DIMC, 0);
  };
  // MLP: consumes LN output in LNB, residual xmbuf, writes outbuf
  auto mlp_block = [&](int mbase, const unsigned short* fc1w, const unsigned short* fc2w,
                       const float* xmbuf, float* outbuf) {
    gemm(LNB, fc1w, F(mbase + 1), nullptr, nullptr, HB, nullptr, HID, DIMC, 1);
    k_dwconv_gelu<<<g((long)LTOT * HID), 256, 0, stream>>>(HB, F(mbase + 2), F(mbase + 3), BB);
    gemm(BB, fc2w, F(mbase + 5), xmbuf, outbuf, nullptr, nullptr, DIMC, HID, 0);
  };

  // ---- global branch (mamba2 / mlp2 / ln3,ln4) ----
  k_pack_xf<<<g((long)LTOT * DIMC), 256, 0, stream>>>(x, XF);
  k_layernorm_bf16<<<LTOT, 128, 0, stream>>>(XF, F(50), F(51), LNB);
  mamba_block(18, XF, BW_INW2, BW_OUTW2, XM);
  k_layernorm_bf16<<<LTOT, 128, 0, stream>>>(XM, F(52), F(53), LNB);
  mlp_block(40, BW_FC1W2, BW_FC2W2, XM, OG);

  // ---- wavelet branch (mamba1 / mlp1 / ln1,ln2) ----
  k_invperm<<<g(LSEQ), 256, 0, stream>>>(sidx, IP);
  k_dwt_gather<<<g((long)LTOT * DIMC), 256, 0, stream>>>(x, sidx, XF);
  k_layernorm_bf16<<<LTOT, 128, 0, stream>>>(XF, F(46), F(47), LNB);
  mamba_block(2, XF, BW_INW1, BW_OUTW1, XM);
  k_layernorm_bf16<<<LTOT, 128, 0, stream>>>(XM, F(48), F(49), LNB);
  mlp_block(34, BW_FC1W1, BW_FC2W1, XM, OM);

  // ---- combine: idwt(scatter(OM)) || OG -> linear 768->384 + residual ----
  k_idwt_cat<<<g((long)LTOT * DIMC), 256, 0, stream>>>(OM, IP, CAT);
  k_cat_global<<<g((long)LTOT * DIMC), 256, 0, stream>>>(OG, CAT);
  gemm(CAT, BW_LIN, F(55), nullptr, (float*)d_out, nullptr, x, DIMC, 2 * DIMC, 4);
#undef F
}
